// TimeModulatedFourierBlock1D_39805756899840
// MI455X (gfx1250) — matrix operand order repaired and checked by reading, not compiled, not run on hardware
//
#include <hip/hip_runtime.h>

// ---------------- problem constants ----------------
#define B_   32
#define G_   4096
#define CI_  64
#define CO_  64
#define TE_  256
#define M_   65          // retained rfft modes
#define MP_  80          // modes padded to 5x16
#define K3_  160         // inverse-basis K: 65 cos + 65 sin + 30 pad
#define EPSV 1e-5f

typedef _Float16 h16;
typedef __attribute__((ext_vector_type(8)))  _Float16 v8h;
typedef __attribute__((ext_vector_type(16))) _Float16 v16h;
typedef __attribute__((ext_vector_type(8)))  float    v8f;

// ---------------- workspace layout ----------------
// f16 region
#define T1_ELEMS   (2*MP_*G_)            // 655360  fwd basis [comp][80][4096]
#define T3_ELEMS   (G_*K3_)              // 655360  inv basis [4096][160]
#define WH_ELEMS   (M_*2*CO_*CI_)        // 532480  spectral weights [n][comp][o][ci]
#define CV_ELEMS   (CO_*CI_)             // 4096    conv weights [o][ci]
#define SP_ELEMS   (B_*CO_*K3_)          // 327680  spec f16 [b][o][160]
#define H16_TOTAL  (T1_ELEMS+T3_ELEMS+WH_ELEMS+CV_ELEMS+SP_ELEMS)   // 2,174,976
// f32 region (starts at byte H16_TOTAL*2 = 4,349,952; 16B aligned)
#define XFT_ELEMS  (2*MP_*B_*CI_)        // 327680  [comp][80][b][ci]
#define TTR_ELEMS  (2*B_*M_)             // 4160    [comp][b][n]
#define ST_ELEMS   128                   // sum[64], sumsq[64]
#define SAB_ELEMS  128                   // scale[64], shift[64]

// ---------------- WMMA fragment helpers (gfx1250 wave32 layouts) ----------------
__device__ __forceinline__ v16h cat8(v8h lo, v8h hi) {
  return __builtin_shufflevector(lo, hi, 0,1,2,3,4,5,6,7,8,9,10,11,12,13,14,15);
}
// A-matrix 16x32 f16: lane half h; elems 0..7 -> K=8h+e, elems 8..15 -> K=16+8h+(e-8)
__device__ __forceinline__ v16h frag_a_h(const h16* row, int h) {
  v8h lo = *(const v8h*)(row + 8*h);
  v8h hi = *(const v8h*)(row + 16 + 8*h);
  return cat8(lo, hi);
}
// B-matrix 32x16 f16: lane half h; elems 0..7 -> K=8h+e, elems 8..15 -> K=16+8h+(e-8)
__device__ __forceinline__ v16h frag_b_h(const h16* row, int h) {
  v8h lo = *(const v8h*)(row + 8*h);
  v8h hi = *(const v8h*)(row + 16 + 8*h);
  return cat8(lo, hi);
}
// A fragment with on-the-fly f32 -> f16 conversion (sources are 32B-aligned)
__device__ __forceinline__ v16h frag_a_f32(const float* row, int h) {
  const float4* p0 = (const float4*)(row + 8*h);
  const float4* p1 = (const float4*)(row + 16 + 8*h);
  float4 a0 = p0[0], a1 = p0[1];
  float4 b0 = p1[0], b1 = p1[1];
  v16h r;
  r[0]  = (h16)a0.x; r[1]  = (h16)a0.y; r[2]  = (h16)a0.z; r[3]  = (h16)a0.w;
  r[4]  = (h16)a1.x; r[5]  = (h16)a1.y; r[6]  = (h16)a1.z; r[7]  = (h16)a1.w;
  r[8]  = (h16)b0.x; r[9]  = (h16)b0.y; r[10] = (h16)b0.z; r[11] = (h16)b0.w;
  r[12] = (h16)b1.x; r[13] = (h16)b1.y; r[14] = (h16)b1.z; r[15] = (h16)b1.w;
  return r;
}
__device__ __forceinline__ v8f wmma16(v16h a, v16h b, v8f c) {
  return __builtin_amdgcn_wmma_f32_16x16x32_f16(false, a, false, b, (short)0, c, false, false);
}

// ---------------- kernel 1: DFT basis tables ----------------
__global__ void k_tables(h16* __restrict__ T1, h16* __restrict__ T3) {
  const float STEP = 6.283185307179586f / (float)G_;
  int total = T1_ELEMS + T3_ELEMS;
  for (int i = blockIdx.x * blockDim.x + threadIdx.x; i < total; i += gridDim.x * blockDim.x) {
    if (i < T1_ELEMS) {               // forward basis: [comp][80][4096], value {cos,-sin}/G
      int comp = i / (MP_*G_);
      int r = i % (MP_*G_);
      int n = r >> 12, g = r & (G_-1);
      float v = 0.f;
      if (n < M_) {
        float th = STEP * (float)((n * g) & (G_-1));   // exact integer phase reduction
        v = ((comp == 0) ? cosf(th) : -sinf(th)) * (1.0f / (float)G_);
      }
      T1[i] = (h16)v;
    } else {                          // inverse basis: [4096][160], c_n {cos,-sin}
      int j = i - T1_ELEMS;
      int g = j / K3_, k = j % K3_;
      float v = 0.f;
      if (k < 2*M_) {
        int n = (k < M_) ? k : (k - M_);
        float c = (n == 0) ? 1.f : 2.f;
        float th = STEP * (float)((n * g) & (G_-1));
        v = (k < M_) ? c * cosf(th) : -c * sinf(th);
      }
      T3[j] = (h16)v;
    }
  }
}

// ---------------- kernel 2: weight conversion + zero accumulators ----------------
__global__ void k_prep(const float* __restrict__ w_real, const float* __restrict__ w_imag,
                       const float* __restrict__ conv_w,
                       h16* __restrict__ wH, h16* __restrict__ cvH, h16* __restrict__ specH,
                       float* __restrict__ xft, float* __restrict__ stats) {
  const int S0 = WH_ELEMS;                  // wH
  const int S1 = S0 + CV_ELEMS;             // convH
  const int S2 = S1 + XFT_ELEMS;            // xft zero
  const int S3 = S2 + ST_ELEMS;             // stats zero
  const int S4 = S3 + B_*CO_*30;            // specH pad zero
  for (int i = blockIdx.x * blockDim.x + threadIdx.x; i < S4; i += gridDim.x * blockDim.x) {
    if (i < S0) {
      int n = i >> 13, r = i & 8191;
      int comp = r >> 12, r2 = r & 4095;
      int o = r2 >> 6, ci = r2 & 63;
      const float* src = comp ? w_imag : w_real;       // (M, CI, CO)
      wH[i] = (h16)src[(n * CI_ + ci) * CO_ + o];
    } else if (i < S1) {
      int j = i - S0; int o = j >> 6, ci = j & 63;
      cvH[j] = (h16)conv_w[ci * CO_ + o];              // (CI, CO) -> [o][ci]
    } else if (i < S2) {
      xft[i - S1] = 0.f;
    } else if (i < S3) {
      stats[i - S2] = 0.f;
    } else {
      int j = i - S3; int t = j / 30; int kk = 2*M_ + (j % 30);
      specH[t * K3_ + kk] = (h16)0.f;                  // pad columns of spec rows
    }
  }
}

// ---------------- kernel 3: t_tr = t_emb @ dense_{r,i} ----------------
__global__ void k_ttr(const float* __restrict__ t_emb, const float* __restrict__ dense_r,
                      const float* __restrict__ dense_i, float* __restrict__ ttr) {
  int id = blockIdx.x * blockDim.x + threadIdx.x;
  if (id >= TTR_ELEMS) return;
  int comp = id / (B_*M_);
  int r = id % (B_*M_);
  int b = r / M_, n = r % M_;
  const float* D = comp ? dense_i : dense_r;           // (TE, M)
  float s = 0.f;
  for (int k = 0; k < TE_; ++k) s += t_emb[b * TE_ + k] * D[k * M_ + n];
  ttr[id] = s;
}

// ---------------- kernel 4: forward DFT GEMM (WMMA, K split across g-chunks) ----------------
#define GC 256                         // g elements per workgroup
#define XS_PITCH 264                   // 256 + 8 pad (16B-aligned rows, de-conflicted banks)
__global__ void __launch_bounds__(320)
k_dft(const float* __restrict__ x, const h16* __restrict__ T1, float* __restrict__ xft) {
  __shared__ h16 xs[CI_ * XS_PITCH];   // x chunk as f16 [ci][g], 33.8 KB
  int b = blockIdx.x >> 4;
  int g0 = (blockIdx.x & 15) << 8;
  int tid = threadIdx.x;
  // stage chunk to LDS (coalesced global reads, f32->f16)
  for (int i = tid; i < CI_ * GC; i += 320) {
    int gg = i >> 6, ci = i & 63;
    xs[ci * XS_PITCH + gg] = (h16)x[((b << 12) + g0 + gg) * CI_ + ci];
  }
  __syncthreads();
  int w = tid >> 5, lane = tid & 31, m = lane & 15, h = lane >> 4;
  int mt = w >> 1, comp = w & 1, nb = mt << 4;          // 10 waves: 5 mode tiles x {re,im}
  const h16* arow = T1 + (((comp * MP_) + nb + m) << 12) + g0;
#pragma unroll
  for (int ct = 0; ct < 4; ++ct) {
    v8f acc = {0.f,0.f,0.f,0.f,0.f,0.f,0.f,0.f};
    const h16* brow = &xs[(ct * 16 + m) * XS_PITCH];
#pragma unroll
    for (int ks = 0; ks < GC/32; ++ks) {
      int kb = ks * 32;
      acc = wmma16(frag_a_h(arow + kb, h), frag_b_h(brow + kb, h), acc);
    }
    int ci = (ct << 4) + m;
    // Unguarded scatter: rows n in [65,80) are allocated, zero-basis -> acc==0,
    // so the atomic adds are harmless and EXEC stays uniform (no saveexec dance).
#pragma unroll
    for (int r = 0; r < 8; ++r) {
      int n = nb + r + (h << 3);
      unsafeAtomicAdd(&xft[(((comp * MP_) + n) * B_ + b) * CI_ + ci], acc[r]);
    }
  }
}

// ---------------- kernel 5: per-mode complex mix x time modulation ----------------
__global__ void __launch_bounds__(256)
k_specmix(const float* __restrict__ xft, const h16* __restrict__ wH,
          const float* __restrict__ ttr, h16* __restrict__ specH) {
  int n = blockIdx.x;                        // 0..64
  int tid = threadIdx.x;
  int w = tid >> 5, lane = tid & 31, m = lane & 15, h = lane >> 4;
  int bt = w >> 2, ot = w & 3;
  const float* xr = xft + ((n)        * B_ + bt * 16 + m) * CI_;
  const float* xi = xft + ((MP_ + n)  * B_ + bt * 16 + m) * CI_;
  const h16*   wr = wH  + ((n * 2 + 0) * CO_ + ot * 16 + m) * CI_;
  const h16*   wi = wH  + ((n * 2 + 1) * CO_ + ot * 16 + m) * CI_;
  v8f Zr = {0.f,0.f,0.f,0.f,0.f,0.f,0.f,0.f};
  v8f Zi = {0.f,0.f,0.f,0.f,0.f,0.f,0.f,0.f};
#pragma unroll
  for (int ks = 0; ks < 2; ++ks) {
    int kb = ks * 32;
    v16h ar = frag_a_f32(xr + kb, h);
    v16h ai = frag_a_f32(xi + kb, h);
    v16h br = frag_b_h(wr + kb, h);
    v16h bi = frag_b_h(wi + kb, h);
    v16h an = -ai;                           // FP16 WMMA NEG bits are C-only -> negate in VALU
    Zr = wmma16(ar, br, Zr);  Zr = wmma16(an, bi, Zr);   // Re = Xr*Wr - Xi*Wi
    Zi = wmma16(ar, bi, Zi);  Zi = wmma16(ai, br, Zi);   // Im = Xr*Wi + Xi*Wr
  }
  int o = ot * 16 + m;
#pragma unroll
  for (int r = 0; r < 8; ++r) {
    int b = bt * 16 + r + (h << 3);
    float tr = ttr[b * M_ + n];
    float ti = ttr[B_*M_ + b * M_ + n];
    h16* dst = specH + (b * CO_ + o) * K3_;
    dst[n]      = (h16)(tr * Zr[r] - ti * Zi[r]);
    dst[M_ + n] = (h16)(tr * Zi[r] + ti * Zr[r]);
  }
}

// ---------------- kernel 6: inverse DFT + residual conv + BN stats ----------------
__global__ void __launch_bounds__(256)
k_idft(const float* __restrict__ x, const h16* __restrict__ T3, const h16* __restrict__ specH,
       const h16* __restrict__ cvH, const float* __restrict__ conv_b,
       float* __restrict__ out, float* __restrict__ stats) {
  __shared__ float lsum[CO_], lss[CO_];
  int b = blockIdx.x >> 7;
  int g0 = (blockIdx.x & 127) << 5;
  int tid = threadIdx.x;
  if (tid < CO_) { lsum[tid] = 0.f; lss[tid] = 0.f; }
  __syncthreads();
  int w = tid >> 5, lane = tid & 31, m = lane & 15, h = lane >> 4;
  int rt = w >> 2, ot = w & 3;
  int gb = g0 + rt * 16;
  const h16* arow = T3 + (gb + m) * K3_;
  const h16* brow = specH + (b * CO_ + ot * 16 + m) * K3_;
  v8f acc = {0.f,0.f,0.f,0.f,0.f,0.f,0.f,0.f};
#pragma unroll
  for (int ks = 0; ks < 5; ++ks) {                        // spectral part, K=160
    int kb = ks * 32;
    acc = wmma16(frag_a_h(arow + kb, h), frag_b_h(brow + kb, h), acc);
  }
  const float* xrow = x + ((b << 12) + gb + m) * CI_;
  const h16* crow = cvH + (ot * 16 + m) * CI_;
#pragma unroll
  for (int ks = 0; ks < 2; ++ks) {                        // residual 1x1 conv, K=64
    int kb = ks * 32;
    acc = wmma16(frag_a_f32(xrow + kb, h), frag_b_h(crow + kb, h), acc);
  }
  int o = ot * 16 + m;
  float bias = conv_b[o];
  float s = 0.f, ss = 0.f;
#pragma unroll
  for (int r = 0; r < 8; ++r) {
    int g = gb + r + (h << 3);
    float y = acc[r] + bias;
    out[((b << 12) + g) * CO_ + o] = y;
    s += y; ss += y * y;
  }
  atomicAdd(&lsum[o], s);                                  // ds_add_f32
  atomicAdd(&lss[o], ss);
  __syncthreads();
  if (tid < CO_) {
    unsafeAtomicAdd(&stats[tid],        lsum[tid]);
    unsafeAtomicAdd(&stats[CO_ + tid],  lss[tid]);
  }
}

// ---------------- kernel 7: finalize BN scale/shift ----------------
__global__ void k_finstats(const float* __restrict__ stats, const float* __restrict__ bn_scale,
                           const float* __restrict__ bn_bias, float* __restrict__ sAB) {
  int o = threadIdx.x;
  if (o >= CO_) return;
  const float cnt = (float)(B_ * G_);
  float mean = stats[o] / cnt;
  float var  = stats[CO_ + o] / cnt - mean * mean;
  float sA = bn_scale[o] * rsqrtf(var + EPSV);
  sAB[o] = sA;
  sAB[CO_ + o] = bn_bias[o] - mean * sA;
}

// ---------------- kernel 8: normalize + ReLU (in place, float4) ----------------
__global__ void k_bnrelu(float* __restrict__ out, const float* __restrict__ sAB) {
  int total4 = (B_ * G_ * CO_) / 4;
  float4* p = (float4*)out;
  for (int i = blockIdx.x * blockDim.x + threadIdx.x; i < total4; i += gridDim.x * blockDim.x) {
    int o = (i * 4) & (CO_ - 1);
    float4 v = p[i];
    v.x = fmaxf(0.f, v.x * sAB[o]     + sAB[CO_ + o]);
    v.y = fmaxf(0.f, v.y * sAB[o + 1] + sAB[CO_ + o + 1]);
    v.z = fmaxf(0.f, v.z * sAB[o + 2] + sAB[CO_ + o + 2]);
    v.w = fmaxf(0.f, v.w * sAB[o + 3] + sAB[CO_ + o + 3]);
    p[i] = v;
  }
}

// ---------------- host launch ----------------
extern "C" void kernel_launch(void* const* d_in, const int* in_sizes, int n_in,
                              void* d_out, int out_size, void* d_ws, size_t ws_size,
                              hipStream_t stream) {
  const float* x        = (const float*)d_in[0];
  const float* t_emb    = (const float*)d_in[1];
  const float* w_real   = (const float*)d_in[2];
  const float* w_imag   = (const float*)d_in[3];
  const float* dense_r  = (const float*)d_in[4];
  const float* dense_i  = (const float*)d_in[5];
  const float* conv_w   = (const float*)d_in[6];
  const float* conv_b   = (const float*)d_in[7];
  const float* bn_scale = (const float*)d_in[8];
  const float* bn_bias  = (const float*)d_in[9];
  float* out = (float*)d_out;

  // workspace carve-up (~5.7 MB total)
  h16* wsh   = (h16*)d_ws;
  h16* T1    = wsh;
  h16* T3    = T1 + T1_ELEMS;
  h16* wH    = T3 + T3_ELEMS;
  h16* cvH   = wH + WH_ELEMS;
  h16* specH = cvH + CV_ELEMS;
  float* wsf   = (float*)((char*)d_ws + (size_t)H16_TOTAL * 2);
  float* xft   = wsf;
  float* ttr   = xft + XFT_ELEMS;
  float* stats = ttr + TTR_ELEMS;
  float* sAB   = stats + ST_ELEMS;

  k_tables  <<<2048, 256, 0, stream>>>(T1, T3);
  k_prep    <<<1024, 256, 0, stream>>>(w_real, w_imag, conv_w, wH, cvH, specH, xft, stats);
  k_ttr     <<<(TTR_ELEMS + 255) / 256, 256, 0, stream>>>(t_emb, dense_r, dense_i, ttr);
  k_dft     <<<B_ * 16, 320, 0, stream>>>(x, T1, xft);
  k_specmix <<<M_, 256, 0, stream>>>(xft, wH, ttr, specH);
  k_idft    <<<B_ * 128, 256, 0, stream>>>(x, T3, specH, cvH, conv_b, out, stats);
  k_finstats<<<1, 64, 0, stream>>>(stats, bn_scale, bn_bias, sAB);
  k_bnrelu  <<<2048, 256, 0, stream>>>(out, sAB);
}
